// LSTMDecoder_27685359190019
// MI455X (gfx1250) — matrix operand order repaired and checked by reading, not compiled, not run on hardware
//
#include <hip/hip_runtime.h>
#include <hip/hip_bf16.h>

// Problem constants (match reference)
#define BSZ    256
#define LATENT 128
#define HID    256   // also K of recurrent GEMM and FEAT
#define FEAT   256
#define G4     1024  // 4*FEAT
#define TSTEPS 512

typedef __attribute__((ext_vector_type(16))) __bf16 v16bf;
typedef __attribute__((ext_vector_type(8)))  float  v8f;

__device__ __forceinline__ unsigned short f2bf(float f) {
    unsigned u = __float_as_uint(f);
    u += 0x7FFFu + ((u >> 16) & 1u);   // round-to-nearest-even
    return (unsigned short)(u >> 16);
}
__device__ __forceinline__ float sigmoidf(float x) {
    return 1.0f / (1.0f + __expf(-x));
}

// ---------------- Prologue 1: x = inputs @ Wd + bd  [256,128]x[128,256] ----
__global__ void dense_x_kernel(const float* __restrict__ inputs,
                               const float* __restrict__ Wd,
                               const float* __restrict__ bd,
                               float* __restrict__ x) {
    const int col = threadIdx.x;   // 0..HID-1
    const int row = blockIdx.x;    // 0..BSZ-1
    float acc = bd[col];
#pragma unroll 4
    for (int k = 0; k < LATENT; ++k)
        acc += inputs[row * LATENT + k] * Wd[k * HID + col];
    x[row * HID + col] = acc;
}

// ---------------- Prologue 2: xz = x @ W + b  [256,256]x[256,1024] ---------
__global__ void xz_kernel(const float* __restrict__ x,
                          const float* __restrict__ W,
                          const float* __restrict__ b,
                          float* __restrict__ xz) {
    const int row = blockIdx.x >> 2;                       // 0..255
    const int col = ((blockIdx.x & 3) << 8) | threadIdx.x; // 0..1023
    float acc = b[col];
#pragma unroll 4
    for (int k = 0; k < HID; ++k)
        acc += x[row * HID + k] * W[k * G4 + col];
    xz[row * G4 + col] = acc;
}

// ---------------- Prologue 3: Ut[n][k] = bf16(U[k][n])  (transpose+convert) -
__global__ void ut_kernel(const float* __restrict__ U,
                          unsigned short* __restrict__ Ut) {
    const int k = threadIdx.x;   // 0..HID-1
    const int n = blockIdx.x;    // 0..G4-1
    Ut[n * HID + k] = f2bf(U[k * G4 + n]);
}

// ---------------- Main LSTM recurrence ------------------------------------
// 16 WGs x 512 threads (16 wave32). WG wg owns batch rows [16wg,16wg+16).
// Wave w owns feature tile [16w,16w+16), computing the 4 gate columns of
// z = xz + h @ U via v_wmma_f32_16x16x32_bf16. h double-buffered in LDS;
// c stays in registers. The B stream (Ut, L2-resident, h-independent) is
// hand-pipelined 2 k-chunks deep with inline-asm global_load_b128 +
// explicit s_wait_loadcnt, and SKEWED across timesteps: chunks 0-1 of step
// t+1 are issued before step t's gate math + barrier, so the per-step
// pipeline-fill L2 round-trip is off the critical chain.
#define HPAD (HID + 8)   // +16B row pad -> lanes hitting distinct LDS banks

union bfrag { uint4 u[2]; v16bf v; };

// One B tile (16 consecutive K bf16 per lane): 2 x b128, clause of 2.
// 'off' must fold to a compile-time constant (callers fully unrolled).
__device__ __forceinline__ void bload(bfrag& f, unsigned uoff,
                                      const unsigned short* Ut, int off) {
    asm volatile("s_clause 0x1\n\t"
                 "global_load_b128 %0, %2, %3 offset:%4\n\t"
                 "global_load_b128 %1, %2, %3 offset:%5"
                 : "=v"(f.u[0]), "=v"(f.u[1])
                 : "v"(uoff), "s"(Ut), "i"(off), "i"(off + 16));
}

// s_wait_loadcnt cnt. Dependency is carried through ONE 32-bit dword of the
// A fragment (only scalar ties lower cleanly on this backend). The WMMA's A
// operand includes the redefined dword, so no WMMA of this k-chunk can be
// scheduled above the wait; all B loads are earlier asm volatiles, so they
// are issued before it.
__device__ __forceinline__ void bwait_a(bfrag& a, int cnt) {
    unsigned w = a.u[0].x;
    asm volatile("s_wait_loadcnt %1" : "+v"(w) : "i"(cnt));
    a.u[0].x = w;
}

__global__ __launch_bounds__(512, 1)
void lstm_kernel(const float* __restrict__ xz,
                 const unsigned short* __restrict__ Ut,
                 float* __restrict__ out) {
    __shared__ __align__(16) unsigned short hbuf[2][16][HPAD]; // ~16.5 KB

    const int wg    = blockIdx.x;          // 0..15
    const int lane  = threadIdx.x & 31;
    const int wv    = threadIdx.x >> 5;    // 0..15
    const int row0  = wg * 16;
    const int laneN = lane & 15;           // tile row (A) / tile col (B,C)
    const int hi    = lane >> 4;           // half-wave selector
    const int nbase = wv * 16;             // feature base for this wave

    // Per-lane byte offset into Ut for this wave's B fragments (g=0, kk=0).
    const unsigned ubase =
        (unsigned)(((nbase + laneN) * HID + 16 * hi) * sizeof(unsigned short));

    // Warm the U stream for this wave's 4 gate rows (L2-resident thereafter)
#pragma unroll
    for (int g = 0; g < 4; ++g)
        __builtin_prefetch((const char*)Ut + ubase + g * (FEAT * HID * 2), 0, 3);

    // h(t=0) = 0 in buffer 0
    for (int i = threadIdx.x; i < 16 * HID; i += blockDim.x)
        hbuf[0][i >> 8][i & 255] = 0;

    // Preload constant xz accumulator tiles (C layout: vgpr r -> row r+8*hi,
    // lane -> col laneN) and zero the cell state.
    v8f xzv[4];
    v8f cv;
#pragma unroll
    for (int g = 0; g < 4; ++g) {
#pragma unroll
        for (int r = 0; r < 8; ++r) {
            const int M = r + 8 * hi;
            xzv[g][r] = xz[(size_t)(row0 + M) * G4 + g * FEAT + nbase + laneN];
        }
    }
#pragma unroll
    for (int r = 0; r < 8; ++r) cv[r] = 0.0f;

    // Force the xz loads to resolve here (input-only asm: no tied operands),
    // so no s_wait_loadcnt 0 appears at first use inside the time loop —
    // that would drain our hand-built B-load pipeline every step.
#pragma unroll
    for (int g = 0; g < 4; ++g) asm volatile("" :: "v"(xzv[g]));

    // Loop-carried B pipeline slots (chunk c lives in slot c%3).
    bfrag B[3][4];

    // Skewed fill for t=0: issue chunks 0 and 1 before entering the loop.
#pragma unroll
    for (int g = 0; g < 4; ++g)
        bload(B[0][g], ubase, Ut, (g * (FEAT * HID) + 0) * 2);
#pragma unroll
    for (int g = 0; g < 4; ++g)
        bload(B[1][g], ubase, Ut, (g * (FEAT * HID) + 32) * 2);

    __syncthreads();

    for (int t = 0; t < TSTEPS; ++t) {
        const int cur = t & 1;
        const int nxt = cur ^ 1;

        v8f acc[4];
#pragma unroll
        for (int kk = 0; kk < HID / 32; ++kk) {
            const int cslot = kk % 3;

            // issue chunk kk+2 before consuming chunk kk
            if (kk + 2 < HID / 32) {
                const int nslot = (kk + 2) % 3;
                const int kb = (kk + 2) * 32;
#pragma unroll
                for (int g = 0; g < 4; ++g)
                    bload(B[nslot][g], ubase, Ut, (g * (FEAT * HID) + kb) * 2);
            }

            // A fragment from LDS (compiler-managed DScnt, short latency)
            const int kbase = kk * 32;
            bfrag a;
            a.u[0] = *(const uint4*)&hbuf[cur][laneN][kbase + 16 * hi];
            a.u[1] = *(const uint4*)&hbuf[cur][laneN][kbase + 16 * hi + 8];

            // LOADcnt budget: chunks kk+1 (8) and kk+2 (8) may stay pending
            if (kk + 2 < HID / 32)      bwait_a(a, 16);
            else if (kk + 1 < HID / 32) bwait_a(a, 8);
            else                        bwait_a(a, 0);

            if (kk == 0) {
                // feed xz directly as the WMMA C operand (no acc init movs)
#pragma unroll
                for (int g = 0; g < 4; ++g)
                    acc[g] = __builtin_amdgcn_wmma_f32_16x16x32_bf16(
                        false, a.v, false, B[cslot][g].v, (short)0, xzv[g],
                        false, false);
            } else {
#pragma unroll
                for (int g = 0; g < 4; ++g)
                    acc[g] = __builtin_amdgcn_wmma_f32_16x16x32_bf16(
                        false, a.v, false, B[cslot][g].v, (short)0, acc[g],
                        false, false);
            }
        }

        // Skewed refill for step t+1: Ut is time-invariant, so issue chunks
        // 0 and 1 NOW — their L2 latency overlaps the sigmoid tail, the LDS
        // h stores and the barrier below. Slots 0,1 were last consumed at
        // kk=6,7 (WAR-safe). Trailing issue at t=TSTEPS-1 is harmless.
#pragma unroll
        for (int g = 0; g < 4; ++g)
            bload(B[0][g], ubase, Ut, (g * (FEAT * HID) + 0) * 2);
#pragma unroll
        for (int g = 0; g < 4; ++g)
            bload(B[1][g], ubase, Ut, (g * (FEAT * HID) + 32) * 2);

        // Gates (keras order i,f,c,o): c = sig(zf)*c + sig(zi)*zg ; h = sig(zo)*c
#pragma unroll
        for (int r = 0; r < 8; ++r) {
            const float zi = acc[0][r], zf = acc[1][r];
            const float zg = acc[2][r], zo = acc[3][r];
            const float cc = sigmoidf(zf) * cv[r] + sigmoidf(zi) * zg;
            cv[r] = cc;
            const float hh = sigmoidf(zo) * cc;
            const int M = r + 8 * hi;
            hbuf[nxt][M][nbase + laneN] = f2bf(hh);
            // write-once output: non-temporal, keep caches for Ut/h traffic
            __builtin_nontemporal_store(
                hh, &out[((size_t)(row0 + M) * TSTEPS + t) * FEAT + nbase + laneN]);
        }
        __syncthreads();   // publish h(t+1) before next step's A loads
    }
}

extern "C" void kernel_launch(void* const* d_in, const int* in_sizes, int n_in,
                              void* d_out, int out_size, void* d_ws, size_t ws_size,
                              hipStream_t stream) {
    const float* inputs = (const float*)d_in[0];
    const float* Wd     = (const float*)d_in[1];
    const float* bd     = (const float*)d_in[2];
    const float* W      = (const float*)d_in[3];
    const float* U      = (const float*)d_in[4];
    const float* b      = (const float*)d_in[5];
    float* out = (float*)d_out;

    char* ws = (char*)d_ws;
    float*          x  = (float*)(ws);                       // 256 KB
    float*          xz = (float*)(ws + 262144);              // 1 MB
    unsigned short* Ut = (unsigned short*)(ws + 262144 + 1048576); // 512 KB

    dense_x_kernel<<<BSZ, HID, 0, stream>>>(inputs, Wd, bd, x);
    xz_kernel<<<BSZ * 4, 256, 0, stream>>>(x, W, b, xz);
    ut_kernel<<<G4, HID, 0, stream>>>(U, Ut);
    lstm_kernel<<<BSZ / 16, 512, 0, stream>>>(xz, Ut, out);
}
